// multiheadAttention_69217692942478
// MI455X (gfx1250) — matrix operand order repaired and checked by reading, not compiled, not run on hardware
//
#include <hip/hip_runtime.h>
#include <hip/hip_fp16.h>

#define DIM_B 4
#define DIM_S 2048
#define DIM_D 1024
#define DIM_H 16
#define DIM_DK 64

typedef __attribute__((ext_vector_type(16))) _Float16 v16h;
typedef __attribute__((ext_vector_type(8)))  float    v8f;

union Frag16 { unsigned int u[8]; v16h h; };

// ---------------- CDNA5 async global->LDS helpers ----------------

__device__ __forceinline__ unsigned lds_off_of(const void* p) {
  // generic LDS addresses carry the LDS byte offset in the low 32 bits
  return (unsigned)(unsigned long long)p;
}

__device__ __forceinline__ void async_ld_b128(unsigned ldsoff, const __half* g) {
  asm volatile("global_load_async_to_lds_b128 %0, %1, off"
               :: "v"(ldsoff), "v"(g) : "memory");
}

template<int N>
__device__ __forceinline__ void wait_async() {
  asm volatile("s_wait_asynccnt %0" :: "i"(N) : "memory");
}

// ---------------- elementwise conversions ----------------

__global__ void cvt_f32_f16_kernel(const float* __restrict__ in,
                                   __half* __restrict__ out, long long n) {
  long long i = (long long)blockIdx.x * blockDim.x + threadIdx.x;
  long long stride = (long long)gridDim.x * blockDim.x;
  for (; i < n; i += stride) out[i] = __float2half(in[i]);
}

// w[H][D][DK] (fp32) -> wt[N=H*DK][D] (f16) : projection weight in NT layout
__global__ void pack_w_kernel(const float* __restrict__ w, __half* __restrict__ wt) {
  long long o = (long long)blockIdx.x * blockDim.x + threadIdx.x; // o = n*D + d
  if (o >= (long long)DIM_D * DIM_D) return;
  int d  = (int)(o % DIM_D);
  int nn = (int)(o / DIM_D);
  int h = nn >> 6, k = nn & 63;
  wt[o] = __float2half(w[(long long)h * DIM_D * DIM_DK + (long long)d * DIM_DK + k]);
}

// in: [B*S, H*DK] (head-major cols)  ->  out: [B][H][DK][S]  (s-contiguous)
__global__ void transpose_heads_kernel(const __half* __restrict__ in,
                                       __half* __restrict__ out) {
  long long o = (long long)blockIdx.x * blockDim.x + threadIdx.x;
  if (o >= (long long)DIM_B * DIM_S * DIM_D) return;
  int s = (int)(o % DIM_S);
  long long r = o / DIM_S;
  int dk = (int)(r & 63);
  int bh = (int)(r >> 6);
  int b = bh >> 4, h = bh & 15;
  out[o] = in[((long long)(b * DIM_S + s)) * DIM_D + h * DIM_DK + dk];
}

// ---------------- generic NT WMMA GEMM, async double-buffered ----------------
// C[m][n] = alpha * sum_k A[m][k]*Bt[n][k]  (+ bias[n])
// A: [M][K] lda, Bt: [N][K] ldb, f16, k-contiguous. Block tile 64 x (32*NT),
// 4 waves (2x2), each wave 32 x (16*NT). Staged K-step = 64 (two 32-deep WMMA
// half-steps per stage). Global->LDS via global_load_async_to_lds_b128,
// double buffered with compile-time buffer indices (manual 2x pipeline unroll).
template<int NT>
__global__ __launch_bounds__(128)
void gemm_nt_wmma(const __half* __restrict__ A, const __half* __restrict__ Bt,
                  const float* __restrict__ bias,
                  __half* __restrict__ outH, float* __restrict__ outF,
                  __half* __restrict__ outHT,
                  int K, int lda, int ldb, int ldc, int ldct,
                  long long sAb, long long sAh, long long sBb, long long sBh,
                  long long sCb, long long sCh, int HH, float alpha)
{
  constexpr int BN  = 32 * NT;     // block tile N
  constexpr int ACH = 4;           // 16B chunks/thread for A tile (64x64 halves)
  constexpr int BCH = NT * 2;      // 16B chunks/thread for B tile (BNx64 halves)
  constexpr int INFLIGHT = ACH + BCH;

  // row stride 72 halves = 144B: 16B aligned, 36-dword banks (conflict-free)
  __shared__ __half As[2][64][72];
  __shared__ __half Bs[2][BN][72];

  const int tid      = threadIdx.x;
  const int lane     = tid & 31;
  const int laneRow  = lane & 15;
  const int laneHalf = lane >> 4;
  const int wave = tid >> 5;
  const int wr = wave >> 1, wc = wave & 1;

  const int bb = blockIdx.z / HH;
  const int hh = blockIdx.z % HH;
  const __half* Ab = A  + (long long)bb * sAb + (long long)hh * sAh;
  const __half* Bb = Bt + (long long)bb * sBb + (long long)hh * sBh;
  const long long offC = (long long)bb * sCb + (long long)hh * sCh;

  const int m0 = blockIdx.y * 64;
  const int n0 = blockIdx.x * BN;

  // hoisted staging descriptors: global base per chunk + LDS offsets (both bufs)
  const __half* gA[ACH]; unsigned dA[2][ACH];
#pragma unroll
  for (int j = 0; j < ACH; ++j) {
    int c = tid + j * 128, r = c >> 3, s = c & 7;     // 8 x 16B per 64-half row
    gA[j] = Ab + (long long)(m0 + r) * lda + s * 8;
    dA[0][j] = lds_off_of(&As[0][r][s * 8]);
    dA[1][j] = lds_off_of(&As[1][r][s * 8]);
  }
  const __half* gB[BCH]; unsigned dB[2][BCH];
#pragma unroll
  for (int j = 0; j < BCH; ++j) {
    int c = tid + j * 128, r = c >> 3, s = c & 7;
    gB[j] = Bb + (long long)(n0 + r) * ldb + s * 8;
    dB[0][j] = lds_off_of(&Bs[0][r][s * 8]);
    dB[1][j] = lds_off_of(&Bs[1][r][s * 8]);
  }

  v8f acc[2][NT] = {};

  auto stageTile = [&](int bufc, int koff) {
#pragma unroll
    for (int j = 0; j < ACH; ++j) async_ld_b128(dA[bufc][j], gA[j] + koff);
#pragma unroll
    for (int j = 0; j < BCH; ++j) async_ld_b128(dB[bufc][j], gB[j] + koff);
  };

  auto computeHalf = [&](int bufc, int kh) {
    const int khoff = kh * 32;
    Frag16 a[2], b[NT];
#pragma unroll
    for (int t = 0; t < 2; ++t) {
      int row = wr * 32 + t * 16 + laneRow;
#pragma unroll
      for (int vv = 0; vv < 8; ++vv) {
        // A k order used here (contiguous, same as B): lanes 0-15 K=0..15, lanes 16-31 K=16..31
        int ka = khoff + (laneHalf << 4) + (vv << 1);
        a[t].u[vv] = *(const unsigned int*)&As[bufc][row][ka];
      }
    }
#pragma unroll
    for (int t = 0; t < NT; ++t) {
      int col = wc * (16 * NT) + t * 16 + laneRow;
#pragma unroll
      for (int vv = 0; vv < 8; ++vv) {
        // B k order used here (contiguous, same as A): lanes 0-15 K=0..15, lanes 16-31 K=16..31
        int kb = khoff + (laneHalf << 4) + (vv << 1);
        b[t].u[vv] = *(const unsigned int*)&Bs[bufc][col][kb];
      }
    }
#pragma unroll
    for (int tm = 0; tm < 2; ++tm)
#pragma unroll
      for (int tn = 0; tn < NT; ++tn)
        acc[tm][tn] = __builtin_amdgcn_wmma_f32_16x16x32_f16(
            false, a[tm].h, false, b[tn].h, (short)0, acc[tm][tn], false, false);
  };

  // ---- software pipeline, 2 stages in flight, literal buffer indices ----
  stageTile(0, 0);
  int k0 = 0;
  for (; k0 + 128 <= K; k0 += 128) {
    // stage s+1 into buf1, consume buf0 (stage s at k0)
    stageTile(1, k0 + 64);
    wait_async<INFLIGHT>();
    __syncthreads();
    computeHalf(0, 0);
    computeHalf(0, 1);
    __syncthreads();
    // stage s+2 into buf0 (if any), consume buf1 (stage s+1 at k0+64)
    if (k0 + 128 < K) {
      stageTile(0, k0 + 128);
      wait_async<INFLIGHT>();
    } else {
      wait_async<0>();
    }
    __syncthreads();
    computeHalf(1, 0);
    computeHalf(1, 1);
    __syncthreads();
  }
  if (k0 < K) {                 // odd stage count: last stage sits in buf0
    wait_async<0>();
    __syncthreads();
    computeHalf(0, 0);
    computeHalf(0, 1);
  }

  // epilogue: C 16x16 f32 layout -> lanes 0-15 hold M=i, lanes 16-31 hold M=8+i
#pragma unroll
  for (int tm = 0; tm < 2; ++tm) {
#pragma unroll
    for (int tn = 0; tn < NT; ++tn) {
      int n = n0 + wc * (16 * NT) + tn * 16 + laneRow;
      float bv = bias ? bias[n] : 0.0f;
#pragma unroll
      for (int i = 0; i < 8; ++i) {
        int m = m0 + wr * 32 + tm * 16 + (laneHalf << 3) + i;
        float val = acc[tm][tn][i] * alpha + bv;
        if (outF)  outF [offC + (long long)m * ldc  + n] = val;
        if (outH)  outH [offC + (long long)m * ldc  + n] = __float2half(val);
        if (outHT) outHT[offC + (long long)n * ldct + m] = __float2half(val);
      }
    }
  }
}

// ---------------- softmax over contiguous 64-element segments, in place ----
__global__ __launch_bounds__(128)
void softmax64_kernel(__half* __restrict__ data) {
  int seg  = blockIdx.x * 4 + (threadIdx.x >> 5);
  int lane = threadIdx.x & 31;
  __half* p = data + (long long)seg * 64 + lane * 2;
  float x0 = __half2float(p[0]);
  float x1 = __half2float(p[1]);
  float mx = fmaxf(x0, x1);
  for (int m = 16; m > 0; m >>= 1) mx = fmaxf(mx, __shfl_xor(mx, m, 32));
  float e0 = __expf(x0 - mx), e1 = __expf(x1 - mx);
  float s = e0 + e1;
  for (int m = 16; m > 0; m >>= 1) s += __shfl_xor(s, m, 32);
  float inv = 1.0f / s;
  p[0] = __float2half(e0 * inv);
  p[1] = __float2half(e1 * inv);
}

// ---------------- orchestration ----------------
extern "C" void kernel_launch(void* const* d_in, const int* in_sizes, int n_in,
                              void* d_out, int out_size, void* d_ws, size_t ws_size,
                              hipStream_t stream) {
  (void)in_sizes; (void)n_in; (void)out_size; (void)ws_size;

  const float* q    = (const float*)d_in[0];
  const float* k    = (const float*)d_in[1];
  const float* v    = (const float*)d_in[2];
  const float* wq_w = (const float*)d_in[3];
  const float* wq_b = (const float*)d_in[4];
  const float* wk_w = (const float*)d_in[5];
  const float* wk_b = (const float*)d_in[6];
  const float* wv_w = (const float*)d_in[7];
  const float* wv_b = (const float*)d_in[8];
  const float* wo_w = (const float*)d_in[9];
  const float* wo_b = (const float*)d_in[10];
  float* out = (float*)d_out;

  const long long P  = (long long)DIM_B * DIM_S * DIM_D;  // 8.4M elems
  const long long W  = (long long)DIM_D * DIM_D;          // 1M elems
  __half* ws   = (__half*)d_ws;
  __half* xq   = ws;            // f16 q            (reused later as attn)
  __half* xk   = xq + P;        // f16 k            (reused later as khT)
  __half* xv   = xk + P;        // f16 v            (reused later as vhT)
  __half* qh   = xv + P;        // q projection
  __half* kh   = qh + P;        // k projection
  __half* vh   = kh + P;        // v projection
  __half* wqt  = vh + P;        // packed weights (NT layout)
  __half* wkt  = wqt + W;
  __half* wvt  = wkt + W;
  __half* wo16 = wvt + W;       // wo_w is already Bt layout for Y = X*wo^T
  __half* mt   = wo16 + W;      // 64 batches of 64x64 (M^T per b,h)
  __half* khT  = xk;
  __half* vhT  = xv;
  __half* attn = xq;

  // 1) fp32 -> f16 conversions + weight packing
  cvt_f32_f16_kernel<<<2048, 256, 0, stream>>>(q, xq, P);
  cvt_f32_f16_kernel<<<2048, 256, 0, stream>>>(k, xk, P);
  cvt_f32_f16_kernel<<<2048, 256, 0, stream>>>(v, xv, P);
  pack_w_kernel<<<(int)(W / 256), 256, 0, stream>>>(wq_w, wqt);
  pack_w_kernel<<<(int)(W / 256), 256, 0, stream>>>(wk_w, wkt);
  pack_w_kernel<<<(int)(W / 256), 256, 0, stream>>>(wv_w, wvt);
  cvt_f32_f16_kernel<<<1024, 256, 0, stream>>>(wo_w, wo16, W);

  const int MM = DIM_B * DIM_S;  // 8192
  dim3 blk(128);

  // 2) per-head projections as one big GEMM each: [8192,1024] x [1024,1024]
  {
    dim3 grid(DIM_D / 128, MM / 64, 1);
    gemm_nt_wmma<4><<<grid, blk, 0, stream>>>(xq, wqt, wq_b, qh, nullptr, nullptr,
        DIM_D, DIM_D, DIM_D, DIM_D, 0,
        0, 0, 0, 0, 0, 0, 1, 1.0f);
    gemm_nt_wmma<4><<<grid, blk, 0, stream>>>(xk, wkt, wk_b, kh, nullptr, nullptr,
        DIM_D, DIM_D, DIM_D, DIM_D, 0,
        0, 0, 0, 0, 0, 0, 1, 1.0f);
    gemm_nt_wmma<4><<<grid, blk, 0, stream>>>(xv, wvt, wv_b, vh, nullptr, nullptr,
        DIM_D, DIM_D, DIM_D, DIM_D, 0,
        0, 0, 0, 0, 0, 0, 1, 1.0f);
  }

  // 3) transpose kh, vh into [B][H][DK][S] for the khT*vh contraction
  transpose_heads_kernel<<<(int)((P + 255) / 256), 256, 0, stream>>>(kh, khT);
  transpose_heads_kernel<<<(int)((P + 255) / 256), 256, 0, stream>>>(vh, vhT);

  // 4) M^T[b,h] = (scale * khT x vhT^T)^T : 64 batches of [64,2048]x[2048,64]
  {
    dim3 grid(1, 1, DIM_B * DIM_H);
    gemm_nt_wmma<2><<<grid, blk, 0, stream>>>(khT, vhT, nullptr, nullptr, nullptr, mt,
        DIM_S, DIM_S, DIM_S, 0, 64,
        (long long)DIM_H * 64 * DIM_S, (long long)64 * DIM_S,
        (long long)DIM_H * 64 * DIM_S, (long long)64 * DIM_S,
        (long long)DIM_H * 64 * 64,    (long long)64 * 64,
        DIM_H, 0.125f /* 1/sqrt(DK) */);
  }

  // 5) attn[b,h] = qh[b,h] x M[b,h]  -> written head-concatenated [B*S, 1024]
  {
    dim3 grid(1, DIM_S / 64, DIM_B * DIM_H);
    gemm_nt_wmma<2><<<grid, blk, 0, stream>>>(qh, mt, nullptr, attn, nullptr, nullptr,
        64, DIM_D, 64, DIM_D, 0,
        (long long)DIM_S * DIM_D, 64,
        (long long)DIM_H * 64 * 64, (long long)64 * 64,
        (long long)DIM_S * DIM_D, 64,
        DIM_H, 1.0f);
  }

  // 6) softmax over each head's 64 outputs (contiguous segments), in place
  softmax64_kernel<<<(MM * DIM_H) / 4, 128, 0, stream>>>(attn);

  // 7) output projection: [8192,1024] x wo^T + wo_b -> fp32 d_out
  {
    dim3 grid(DIM_D / 128, MM / 64, 1);
    gemm_nt_wmma<4><<<grid, blk, 0, stream>>>(attn, wo16, wo_b, nullptr, out, nullptr,
        DIM_D, DIM_D, DIM_D, DIM_D, 0,
        0, 0, 0, 0, 0, 0, 1, 1.0f);
  }
}
